// MocoLossModule_72825465471100
// MI455X (gfx1250) — matrix operand order repaired and checked by reading, not compiled, not run on hardware
//
#include <hip/hip_runtime.h>

#define BATCH 256
#define DIM   128
#define KNEG  4096
#define NTILE (KNEG / 16)     // 256 k-tiles of 16 per batch row
#define WAVES 8
#define TPW   (NTILE / WAVES) // 32 tiles per wave
#define T_INV (1.0f / 0.07f)
#define EPS   1e-12f

typedef __attribute__((ext_vector_type(16))) _Float16 v16h;
typedef __attribute__((ext_vector_type(8)))  float    v8f;

__device__ __forceinline__ float wave_sum(float v) {
#pragma unroll
  for (int m = 16; m >= 1; m >>= 1) v += __shfl_xor(v, m, 32);
  return v;
}
__device__ __forceinline__ float wave_max(float v) {
#pragma unroll
  for (int m = 16; m >= 1; m >>= 1) v = fmaxf(v, __shfl_xor(v, m, 32));
  return v;
}

// ---------------------------------------------------------------------------
// Kernel A: normalize q,k (one wave per batch row); emit kf to output,
// q as split-f16 (hi + residual) for the WMMA path, and data_logit = q.kf
// ---------------------------------------------------------------------------
__global__ __launch_bounds__(32) void moco_normalize_kernel(
    const float* __restrict__ fea_q, const float* __restrict__ fea_k,
    float* __restrict__ kf_out,                 // d_out + 1
    unsigned short* __restrict__ qh, unsigned short* __restrict__ ql,
    float* __restrict__ dl) {
  const int b = blockIdx.x, lane = threadIdx.x;
  const float4 fq = ((const float4*)(fea_q + (size_t)b * DIM))[lane];
  const float4 fk = ((const float4*)(fea_k + (size_t)b * DIM))[lane];

  float sq = fq.x * fq.x + fq.y * fq.y + fq.z * fq.z + fq.w * fq.w;
  float sk = fk.x * fk.x + fk.y * fk.y + fk.z * fk.z + fk.w * fk.w;
  const float nq = sqrtf(wave_sum(sq));
  const float nk = sqrtf(wave_sum(sk));
  const float cq = 1.0f / fmaxf(nq, EPS);
  const float ck = 1.0f / fmaxf(nk, EPS);

  float q[4] = {fq.x * cq, fq.y * cq, fq.z * cq, fq.w * cq};
  float kk[4] = {fk.x * ck, fk.y * ck, fk.z * ck, fk.w * ck};

  const int d0 = lane * 4;
  float dot = 0.0f;
#pragma unroll
  for (int j = 0; j < 4; ++j) {
    kf_out[(size_t)b * DIM + d0 + j] = kk[j];
    _Float16 h = (_Float16)q[j];
    qh[(size_t)b * DIM + d0 + j] = __builtin_bit_cast(unsigned short, h);
    _Float16 l = (_Float16)(q[j] - (float)h);
    ql[(size_t)b * DIM + d0 + j] = __builtin_bit_cast(unsigned short, l);
    dot += q[j] * kk[j];
  }
  const float dt = wave_sum(dot);
  if (lane == 0) dl[b] = dt;
}

// ---------------------------------------------------------------------------
// Kernel B: per batch row — gather bank rows, WMMA split-f16 GEMV into LDS,
// then fused logsumexp over {data_logit} ∪ 4096 noise logits (all / T).
// Three independent accumulators (hi*hi, hi*lo, lo*hi) so adjacent WMMAs
// never hit the D->C RAW hazard window (no v_nop stalls on the matrix pipe).
// ---------------------------------------------------------------------------
__global__ __launch_bounds__(256) void moco_noise_lse_kernel(
    const float* __restrict__ bank, const int* __restrict__ noise_indx,
    const unsigned short* __restrict__ qh, const unsigned short* __restrict__ ql,
    const float* __restrict__ dl, float* __restrict__ lse_out) {
  __shared__ float s_logits[KNEG];
  __shared__ float s_red[WAVES];
  __shared__ float s_bcast;

  const int b = blockIdx.x;
  const int t = threadIdx.x;
  const int w = t >> 5;
  const int lane = t & 31;
  const int m = lane & 15;   // A-matrix row within tile
  const int hi = lane >> 4;  // lane group (0: K-off 0..15, 1: 16..31)

  // B operand: q_b replicated across all 16 columns; lane group g holds K rows
  // (32*s + 16*g) .. +15 per WMMA step s. Keep hi & lo halves resident.
  v16h bh[4], bl[4];
  {
    const unsigned short* qhb = qh + (size_t)b * DIM;
    const unsigned short* qlb = ql + (size_t)b * DIM;
#pragma unroll
    for (int s = 0; s < 4; ++s) {
      const int base = 32 * s + hi * 16;
      bh[s] = *(const v16h*)(qhb + base);
      bl[s] = *(const v16h*)(qlb + base);
    }
  }

  const int* nrow = noise_indx + (size_t)b * KNEG;

  for (int i = 0; i < TPW; ++i) {
    const int kt = w * TPW + i;
    const int row = nrow[kt * 16 + m];
    const float4* src = (const float4*)(bank + (size_t)row * DIM);

    if (i + 1 < TPW) {  // hide next gather behind this tile's WMMAs
      const int nr = nrow[(kt + 1) * 16 + m];
      __builtin_prefetch(bank + (size_t)nr * DIM + hi * 8, 0, 1);
    }

    v8f c1 = {0.f, 0.f, 0.f, 0.f, 0.f, 0.f, 0.f, 0.f};  // hi*hi
    v8f c2 = {0.f, 0.f, 0.f, 0.f, 0.f, 0.f, 0.f, 0.f};  // hi*lo
    v8f c3 = {0.f, 0.f, 0.f, 0.f, 0.f, 0.f, 0.f, 0.f};  // lo*hi
#pragma unroll
    for (int s = 0; s < 4; ++s) {
      // A layout (16-bit, 16x32): lane (m, hi) holds d = 32s+16*hi+0..15 in
      // elements 0..15 (contiguous, matching B).
      const int q0 = 8 * s + hi * 4;  // float4 index
      const float4 f0 = src[q0], f1 = src[q0 + 1];
      const float4 f2 = src[q0 + 2], f3 = src[q0 + 3];
      const float xf[16] = {f0.x, f0.y, f0.z, f0.w, f1.x, f1.y, f1.z, f1.w,
                            f2.x, f2.y, f2.z, f2.w, f3.x, f3.y, f3.z, f3.w};
      v16h ah, al;
#pragma unroll
      for (int e = 0; e < 16; ++e) {
        const float x = xf[e];
        const _Float16 h = (_Float16)x;
        ah[e] = h;
        al[e] = (_Float16)(x - (float)h);  // residual: fp32-class accuracy
      }
      c1 = __builtin_amdgcn_wmma_f32_16x16x32_f16(false, ah, false, bh[s],
                                                  (short)0, c1, false, false);
      c2 = __builtin_amdgcn_wmma_f32_16x16x32_f16(false, ah, false, bl[s],
                                                  (short)0, c2, false, false);
      c3 = __builtin_amdgcn_wmma_f32_16x16x32_f16(false, al, false, bh[s],
                                                  (short)0, c3, false, false);
    }
    // D: VGPR r, lane L -> M = r + 8*(L>>4), all 16 columns identical.
    if (m == 0) {
      const int kb = kt * 16 + hi * 8;
#pragma unroll
      for (int r = 0; r < 8; ++r) s_logits[kb + r] = c1[r] + (c2[r] + c3[r]);
    }
  }
  __syncthreads();

  // Fused logsumexp over 1 + 4096 logits (scaled by 1/T).
  const float d0 = dl[b] * T_INV;
  float vals[16];
  float lmax = (t == 0) ? d0 : -3.4e38f;
#pragma unroll
  for (int j = 0; j < 16; ++j) {
    vals[j] = s_logits[t * 16 + j] * T_INV;
    lmax = fmaxf(lmax, vals[j]);
  }
  const float wmx = wave_max(lmax);
  if (lane == 0) s_red[w] = wmx;
  __syncthreads();
  if (t == 0) {
    float mm = s_red[0];
#pragma unroll
    for (int j = 1; j < WAVES; ++j) mm = fmaxf(mm, s_red[j]);
    s_bcast = mm;
  }
  __syncthreads();
  const float bmax = s_bcast;

  float lsum = (t == 0) ? __expf(d0 - bmax) : 0.0f;
#pragma unroll
  for (int j = 0; j < 16; ++j) lsum += __expf(vals[j] - bmax);
  const float wsm = wave_sum(lsum);
  __syncthreads();
  if (lane == 0) s_red[w] = wsm;
  __syncthreads();
  if (t == 0) {
    float ss = 0.0f;
#pragma unroll
    for (int j = 0; j < WAVES; ++j) ss += s_red[j];
    lse_out[b] = __logf(ss) + bmax - d0;  // logsumexp - logits[:,0]
  }
}

// ---------------------------------------------------------------------------
// Kernel C: mean over batch -> loss scalar
// ---------------------------------------------------------------------------
__global__ __launch_bounds__(256) void moco_mean_kernel(
    const float* __restrict__ lse, float* __restrict__ out0) {
  __shared__ float s_red[WAVES];
  const int t = threadIdx.x, w = t >> 5, lane = t & 31;
  const float v = wave_sum(lse[t]);
  if (lane == 0) s_red[w] = v;
  __syncthreads();
  if (t == 0) {
    float s = 0.0f;
#pragma unroll
    for (int j = 0; j < WAVES; ++j) s += s_red[j];
    out0[0] = s * (1.0f / (float)BATCH);
  }
}

extern "C" void kernel_launch(void* const* d_in, const int* in_sizes, int n_in,
                              void* d_out, int out_size, void* d_ws, size_t ws_size,
                              hipStream_t stream) {
  // inputs: idx (unused), fea_q, fea_k, bank, noise_indx
  const float* fea_q = (const float*)d_in[1];
  const float* fea_k = (const float*)d_in[2];
  const float* bank  = (const float*)d_in[3];
  const int*   noise = (const int*)d_in[4];
  float* out = (float*)d_out;  // [0]=loss, [1..32768]=kf

  char* ws = (char*)d_ws;
  float* dl  = (float*)ws;                       // 256 f32  data logits
  float* lse = (float*)(ws + 1024);              // 256 f32  per-row loss terms
  unsigned short* qh = (unsigned short*)(ws + 2048);                    // q hi f16
  unsigned short* ql = (unsigned short*)(ws + 2048 + BATCH * DIM * 2);  // q lo f16

  moco_normalize_kernel<<<BATCH, 32, 0, stream>>>(fea_q, fea_k, out + 1, qh, ql, dl);
  moco_noise_lse_kernel<<<BATCH, 256, 0, stream>>>(bank, noise, qh, ql, dl, lse);
  moco_mean_kernel<<<1, 256, 0, stream>>>(lse, out);
}
